// ConvAttention2D_44135083933952
// MI455X (gfx1250) — matrix operand order repaired and checked by reading, not compiled, not run on hardware
//
#include <hip/hip_runtime.h>

typedef _Float16 v16h __attribute__((ext_vector_type(16)));
typedef _Float16 v8h  __attribute__((ext_vector_type(8)));
typedef float    v8f  __attribute__((ext_vector_type(8)));
typedef int      v4i  __attribute__((ext_vector_type(4)));

#define DEV __device__ __forceinline__

#if defined(__AMDGCN__) && __has_builtin(__builtin_amdgcn_global_load_async_to_lds_b128) && __has_builtin(__builtin_amdgcn_s_wait_asynccnt)
#define ASYNC_OK 1
#else
#define ASYNC_OK 0
#endif

constexpr int B_  = 16;
constexpr int E_  = 16;
constexpr int T_  = 1024;
constexpr int F_  = 80;
constexpr int H_  = 8;
constexpr int FP_ = 96;             // F padded to 3 * 32 for WMMA K-dim
constexpr int BH_ = B_ * H_;        // 128
constexpr int C3_ = 3 * H_;         // 24 qkv channels
constexpr int TF_ = T_ * F_;        // 81920
constexpr int NBLK_ = TF_ / 256;    // 320 blocks per (b, oc) in conv kernels

// ---------------------------------------------------------------- helpers ---

DEV v8f vzero8() {
  v8f z;
  #pragma unroll
  for (int i = 0; i < 8; ++i) z[i] = 0.f;
  return z;
}

DEV v8f vscale8(v8f a, float s) {
  #pragma unroll
  for (int i = 0; i < 8; ++i) a[i] *= s;
  return a;
}

// 16B global -> LDS copy; async (ASYNCcnt-tracked) when available.
DEV void cp16(void* lds, const void* g) {
#if ASYNC_OK
  void* gnc = const_cast<void*>(g);
  __builtin_amdgcn_global_load_async_to_lds_b128(
      (__attribute__((address_space(1))) v4i*)gnc,
      (__attribute__((address_space(3))) v4i*)lds, 0, 0);
#else
  *(float4*)lds = *(const float4*)g;
#endif
}

DEV void cp_wait() {
#if ASYNC_OK
  __builtin_amdgcn_s_wait_asynccnt(0);
#endif
}

// A-operand (16x32 f16): lane l (m = l&15, hg = l>>4) holds K = hg*16 + j,
// i.e. 16 contiguous halves (32B).
DEV v16h load_a16(const _Float16* p, int hg) {
  return *(const v16h*)(p + hg * 16);
}

// B-operand (32x16 f16): lane l (n = l&15, hg = l>>4) holds K = hg*16 + j,
// i.e. 16 contiguous halves (32B) of the "row n, K contiguous" view.
DEV v16h load_b16(const _Float16* p, int hg) {
  return *(const v16h*)(p + hg * 16);
}

DEV v8f wmma16(v16h a, v16h b, v8f c) {
  return __builtin_amdgcn_wmma_f32_16x16x32_f16(
      /*neg_a=*/false, a, /*neg_b=*/false, b,
      /*c_mod=*/(short)0, c, /*reuse_a=*/false, /*reuse_b=*/false);
}

// ------------------------------------------------------ conv3x3 + stats -----
// One thread per output element; per-block partial sum/sumsq for BN stats.
// grid = (NBLK_, B_*Cout), block = 256.
__global__ __launch_bounds__(256)
void k_conv3x3_stats(const float* __restrict__ X, const float* __restrict__ W,
                     const float* __restrict__ Bias, float* __restrict__ Y,
                     float* __restrict__ partS, float* __restrict__ partQ,
                     int Cin, int Cout, int scaleLim) {
  const int boc = blockIdx.y;
  const int b = boc / Cout, oc = boc % Cout;
  const int tid = threadIdx.x;

  __shared__ float sW[16 * 9];
  const int nW = Cin * 9;
  for (int i = tid; i < nW; i += 256) sW[i] = W[(size_t)oc * nW + i];
  __syncthreads();

  const int idx = blockIdx.x * 256 + tid;   // 0 .. TF_-1
  const int t = idx / F_, f = idx % F_;

  float acc = Bias[oc];
  for (int ic = 0; ic < Cin; ++ic) {
    const float* xp = X + (((size_t)b * Cin + ic) * T_ + t) * F_ + f;
    const float* wp = sW + ic * 9;
    #pragma unroll
    for (int kh = -1; kh <= 1; ++kh) {
      if ((unsigned)(t + kh) < (unsigned)T_) {
        #pragma unroll
        for (int kw = -1; kw <= 1; ++kw) {
          if ((unsigned)(f + kw) < (unsigned)F_)
            acc += wp[(kh + 1) * 3 + (kw + 1)] * xp[kh * F_ + kw];
        }
      }
    }
  }
  if (oc < scaleLim) acc *= 0.25f;          // q scaling (E^-0.5) pre-BN
  Y[((size_t)boc) * TF_ + idx] = acc;

  __shared__ float rs[256], rq[256];
  rs[tid] = acc; rq[tid] = acc * acc;
  __syncthreads();
  for (int s = 128; s > 0; s >>= 1) {
    if (tid < s) { rs[tid] += rs[tid + s]; rq[tid] += rq[tid + s]; }
    __syncthreads();
  }
  if (tid == 0) {
    const size_t pi = (size_t)oc * (B_ * NBLK_) + b * NBLK_ + blockIdx.x;
    partS[pi] = rs[0];
    partQ[pi] = rq[0];
  }
}

// ------------------------------------------------- BN finalize (in / out) ---
DEV void bn_reduce(const float* pS, const float* pQ, int c, int np,
                   float invN, float gamma, float beta, float* stats) {
  const int tid = threadIdx.x;
  __shared__ float rs[256], rq[256];
  float s = 0.f, q = 0.f;
  for (int i = tid; i < np; i += 256) {
    s += pS[(size_t)c * np + i];
    q += pQ[(size_t)c * np + i];
  }
  rs[tid] = s; rq[tid] = q;
  __syncthreads();
  for (int k = 128; k > 0; k >>= 1) {
    if (tid < k) { rs[tid] += rs[tid + k]; rq[tid] += rq[tid + k]; }
    __syncthreads();
  }
  if (tid == 0) {
    const float mean = rs[0] * invN;
    const float var  = rq[0] * invN - mean * mean;
    const float sc   = gamma * rsqrtf(var + 1e-5f);
    stats[2 * c]     = sc;
    stats[2 * c + 1] = beta - mean * sc;
  }
}

__global__ __launch_bounds__(256)
void k_bn_finalize_in(const float* pS, const float* pQ,
                      const float* gq, const float* bq,
                      const float* gk, const float* bk,
                      const float* gv, const float* bv, float* stats) {
  const int c = blockIdx.x;                 // 0..23
  const float* g = (c < 8) ? gq : (c < 16) ? gk : gv;
  const float* bb = (c < 8) ? bq : (c < 16) ? bk : bv;
  bn_reduce(pS, pQ, c, B_ * NBLK_, 1.f / (float)(B_ * TF_), g[c & 7], bb[c & 7], stats);
}

__global__ __launch_bounds__(256)
void k_bn_finalize_out(const float* pS, const float* pQ,
                       const float* go, const float* bo, float* stats) {
  const int c = blockIdx.x;                 // 0..15
  bn_reduce(pS, pQ, c, B_ * NBLK_, 1.f / (float)(B_ * TF_), go[c], bo[c], stats);
}

// ------------------------------------- BN+ReLU, f16 convert, 6 layouts ------
// Row-major Q16/K16/V16: [bh][T][FP]; transposed Qt/Kt/Vt: [bh][FP][T].
__global__ __launch_bounds__(256)
void k_pack(const float* __restrict__ qkv, const float* __restrict__ stats,
            _Float16* __restrict__ Q16, _Float16* __restrict__ K16,
            _Float16* __restrict__ V16, _Float16* __restrict__ Qt,
            _Float16* __restrict__ Kt, _Float16* __restrict__ Vt) {
  const size_t gid = (size_t)blockIdx.x * 256 + threadIdx.x;  // BH_*T_*FP_
  const int f = (int)(gid % FP_);
  const size_t r = gid / FP_;
  const int t = (int)(r % T_);
  const int bh = (int)(r / T_);
  const int b = bh >> 3, hh = bh & 7;

  float q = 0.f, k = 0.f, v = 0.f;
  if (f < F_) {
    const size_t base = (((size_t)b * C3_) * T_ + t) * F_ + f;
    const float qv = qkv[base + (size_t)hh * TF_];
    const float kv = qkv[base + (size_t)(8 + hh) * TF_];
    const float vv = qkv[base + (size_t)(16 + hh) * TF_];
    q = fmaxf(qv * stats[2 * hh] + stats[2 * hh + 1], 0.f);
    k = fmaxf(kv * stats[2 * (8 + hh)] + stats[2 * (8 + hh) + 1], 0.f);
    v = fmaxf(vv * stats[2 * (16 + hh)] + stats[2 * (16 + hh) + 1], 0.f);
  }
  const _Float16 qh = (_Float16)q, kh = (_Float16)k, vh = (_Float16)v;
  const size_t rm = ((size_t)bh * T_ + t) * FP_ + f;
  const size_t cm = ((size_t)bh * FP_ + f) * T_ + t;
  Q16[rm] = qh; K16[rm] = kh; V16[rm] = vh;
  Qt[cm] = qh;  Kt[cm] = kh;  Vt[cm] = vh;
}

// --------------------------------------------- time attention (flash) -------
// One wave per (bh, 16-query tile); 4 waves per block share bh, so K/Vt
// chunks are cooperatively staged (async) into double-buffered LDS.
// Scores computed transposed (C[m=s][n=q]) so flash softmax is per-lane
// + one shfl_xor(16). grid = (BH_, 16), block = 128 (4 waves).
__global__ __launch_bounds__(128)
void k_time_attn(const _Float16* __restrict__ Q16, const _Float16* __restrict__ K16,
                 const _Float16* __restrict__ Vt16, float* __restrict__ attn,
                 float* __restrict__ ml) {
  const int tid  = threadIdx.x;
  const int lane = tid & 31;
  const int w    = tid >> 5;
  const int nq   = lane & 15;
  const int hg   = lane >> 4;
  const int bh   = blockIdx.x;
  const int qbase = (blockIdx.y * 4 + w) * 16;

  __shared__ _Float16 sK[2][32][FP_];       // 12 KB: 32 keys x 96 (padded F)
  __shared__ _Float16 sV[2][80][32];        // 10 KB: Vt rows x 32 keys
  __shared__ _Float16 pt[4][16][32];        // per-wave P^T staging (B-layout)

  const _Float16* Kbase = K16 + (size_t)bh * T_ * FP_;
  const _Float16* Vbase = Vt16 + (size_t)bh * FP_ * T_;

  auto stage = [&](int buf, int sb) {
    const _Float16* gk = Kbase + (size_t)sb * FP_;   // contiguous 6144 B
    #pragma unroll
    for (int i = 0; i < 3; ++i) {
      const int idx = tid + i * 128;                 // 0..383 (x16B)
      cp16(&sK[buf][0][0] + idx * 8, gk + idx * 8);
    }
    #pragma unroll
    for (int i = 0; i < 3; ++i) {
      const int idx = tid + i * 128;                 // 0..319 valid (x16B)
      if (idx < 320) {
        const int r = idx >> 2, c = (idx & 3) * 8;
        cp16(&sV[buf][r][c], Vbase + (size_t)r * T_ + sb + c);
      }
    }
  };

  // B operand: this wave's Q rows (kept in registers for all chunks)
  const _Float16* qrow = Q16 + ((size_t)bh * T_ + qbase + nq) * FP_;
  v16h Bq[3];
  #pragma unroll
  for (int kb = 0; kb < 3; ++kb) Bq[kb] = load_b16(qrow + kb * 32, hg);

  float m = -1e30f, l = 0.f;
  v8f acc[5];
  #pragma unroll
  for (int ft = 0; ft < 5; ++ft) acc[ft] = vzero8();

  stage(0, 0);
  cp_wait();
  __syncthreads();

  for (int c = 0; c < T_ / 32; ++c) {
    const int buf = c & 1;
    if (c + 1 < T_ / 32) stage(buf ^ 1, (c + 1) * 32);

    // ---- transposed score tiles from LDS-staged K chunk.
    // Load ALL A-tiles into distinct registers first so the in-order DS
    // returns pipeline (partial dscnt waits) instead of one round-trip
    // per WMMA.
    const _Float16* k0 = &sK[buf][nq][0];
    const _Float16* k1 = &sK[buf][16 + nq][0];
    v16h a0[3], a1[3];
    #pragma unroll
    for (int kb = 0; kb < 3; ++kb) {
      a0[kb] = load_a16(k0 + kb * 32, hg);
      a1[kb] = load_a16(k1 + kb * 32, hg);
    }
    v8f c0 = vzero8(), c1 = vzero8();
    #pragma unroll
    for (int kb = 0; kb < 3; ++kb) {
      c0 = wmma16(a0[kb], Bq[kb], c0);
      c1 = wmma16(a1[kb], Bq[kb], c1);
    }

    // ---- online softmax (per query column == per lane pair)
    float cm = -1e30f;
    #pragma unroll
    for (int r = 0; r < 8; ++r) cm = fmaxf(cm, fmaxf(c0[r], c1[r]));
    cm = fmaxf(cm, __shfl_xor(cm, 16));
    const float mnew = fmaxf(m, cm);
    const float corr = __expf(m - mnew);
    float p0[8], p1[8], ls = 0.f;
    #pragma unroll
    for (int r = 0; r < 8; ++r) {
      p0[r] = __expf(c0[r] - mnew);
      p1[r] = __expf(c1[r] - mnew);
      ls += p0[r] + p1[r];
    }
    ls += __shfl_xor(ls, 16);
    l = l * corr + ls;
    #pragma unroll
    for (int ft = 0; ft < 5; ++ft) acc[ft] = vscale8(acc[ft], corr);

    // ---- C-layout -> B-operand layout via wave-private LDS
    v8h h0, h1;
    #pragma unroll
    for (int r = 0; r < 8; ++r) { h0[r] = (_Float16)p0[r]; h1[r] = (_Float16)p1[r]; }
    *(v8h*)&pt[w][nq][hg * 8]      = h0;    // s = hg*8 + r
    *(v8h*)&pt[w][nq][16 + hg * 8] = h1;    // s = 16 + hg*8 + r
    const v16h pb = *(const v16h*)&pt[w][nq][hg * 16];

    // ---- O^T += Vt_tile(A) * P^T(B), K = 32 keys of this chunk
    v16h av[5];
    #pragma unroll
    for (int ft = 0; ft < 5; ++ft)
      av[ft] = load_a16(&sV[buf][ft * 16 + nq][0], hg);
    #pragma unroll
    for (int ft = 0; ft < 5; ++ft)
      acc[ft] = wmma16(av[ft], pb, acc[ft]);
    m = mnew;

    cp_wait();
    __syncthreads();
  }

  const float rl = 1.f / l;
  const int b = bh >> 3, hh = bh & 7;
  const int t = qbase + nq;
  #pragma unroll
  for (int ft = 0; ft < 5; ++ft) {
    float* op = attn + (((size_t)(b * 2 * H_ + hh) * T_ + t) * F_) + ft * 16 + hg * 8;
    #pragma unroll
    for (int r = 0; r < 8; ++r) op[r] = acc[ft][r] * rl;
  }
  if (hg == 0) {
    float2 mv; mv.x = m; mv.y = l;
    *(float2*)&ml[2 * ((size_t)bh * T_ + t)] = mv;
  }
}

// ------------------------------------------------ head-averaged aw_t --------
// One wave per (b, q-tile); serially accumulates 8 heads (deterministic).
// The 4 waves share b, so the per-head K chunks are staged cooperatively.
// grid = (B_, 16), block = 128.
__global__ __launch_bounds__(128)
void k_aw(const _Float16* __restrict__ Q16, const _Float16* __restrict__ K16,
          const float* __restrict__ ml, float* __restrict__ aw) {
  const int tid  = threadIdx.x;
  const int lane = tid & 31;
  const int w    = tid >> 5;
  const int nq   = lane & 15;
  const int hg   = lane >> 4;
  const int b    = blockIdx.x;
  const int qbase = (blockIdx.y * 4 + w) * 16;
  const int t = qbase + nq;

  __shared__ _Float16 sK[2][32][FP_];

  auto stageK = [&](int buf, int j) {       // j = hh*32 + chunk
    const int hh = j >> 5, sb = (j & 31) * 32;
    const _Float16* gk = K16 + ((size_t)(b * H_ + hh) * T_ + sb) * FP_;
    #pragma unroll
    for (int i = 0; i < 3; ++i) {
      const int idx = tid + i * 128;
      cp16(&sK[buf][0][0] + idx * 8, gk + idx * 8);
    }
  };

  stageK(0, 0);
  cp_wait();
  __syncthreads();

  int j = 0;
  for (int hh = 0; hh < H_; ++hh) {
    const int bh = b * H_ + hh;
    const _Float16* qrow = Q16 + ((size_t)bh * T_ + t) * FP_;
    v16h Bq[3];
    #pragma unroll
    for (int kb = 0; kb < 3; ++kb) Bq[kb] = load_b16(qrow + kb * 32, hg);
    const float mq  = ml[2 * ((size_t)bh * T_ + t)];
    const float scl = 0.125f / ml[2 * ((size_t)bh * T_ + t) + 1];

    for (int sbi = 0; sbi < 32; ++sbi) {
      const int buf = j & 1;
      if (j + 1 < H_ * 32) stageK(buf ^ 1, j + 1);

      const int sb = sbi * 32;
      const _Float16* k0 = &sK[buf][nq][0];
      const _Float16* k1 = &sK[buf][16 + nq][0];
      v16h a0[3], a1[3];
      #pragma unroll
      for (int kb = 0; kb < 3; ++kb) {
        a0[kb] = load_a16(k0 + kb * 32, hg);
        a1[kb] = load_a16(k1 + kb * 32, hg);
      }
      v8f c0 = vzero8(), c1 = vzero8();
      #pragma unroll
      for (int kb = 0; kb < 3; ++kb) {
        c0 = wmma16(a0[kb], Bq[kb], c0);
        c1 = wmma16(a1[kb], Bq[kb], c1);
      }
      float* ap = aw + ((size_t)b * T_ + t) * T_ + sb + hg * 8;
      if (hh == 0) {
        #pragma unroll
        for (int r = 0; r < 8; ++r) {
          ap[r]      = __expf(c0[r] - mq) * scl;
          ap[16 + r] = __expf(c1[r] - mq) * scl;
        }
      } else {
        #pragma unroll
        for (int r = 0; r < 8; ++r) {
          ap[r]      += __expf(c0[r] - mq) * scl;
          ap[16 + r] += __expf(c1[r] - mq) * scl;
        }
      }
      cp_wait();
      __syncthreads();
      ++j;
    }
  }
}

// ------------------------------------------------ frequency attention -------
// One wave per (bh, 16-row f tile). Scores C[m=f][n=g] over K = T;
// softmax over g (5 live tiles, pad tile fully masked).
// grid = (BH_, 5), block = 32.
__global__ __launch_bounds__(32)
void k_freq_attn(const _Float16* __restrict__ Qt, const _Float16* __restrict__ Kt,
                 const _Float16* __restrict__ V16, float* __restrict__ attn) {
  const int lane = threadIdx.x;
  const int nq = lane & 15;
  const int hg = lane >> 4;
  const int bh = blockIdx.x;
  const int ftile = blockIdx.y;

  __shared__ _Float16 pl[16][FP_];

  v8f sc[5];
  #pragma unroll
  for (int gt = 0; gt < 5; ++gt) sc[gt] = vzero8();

  const _Float16* qrow = Qt + ((size_t)bh * FP_ + ftile * 16 + nq) * T_;
  const _Float16* Kb   = Kt + (size_t)bh * FP_ * T_;

  for (int kb = 0; kb < T_; kb += 32) {
    const v16h a = load_a16(qrow + kb, hg);
    #pragma unroll
    for (int gt = 0; gt < 5; ++gt) {
      const v16h bk = load_b16(Kb + (size_t)(gt * 16 + nq) * T_ + kb, hg);
      sc[gt] = wmma16(a, bk, sc[gt]);
    }
  }

  // row softmax over g in [0,80): reduce across the 16 lanes of this half.
  #pragma unroll
  for (int r = 0; r < 8; ++r) {
    float mx = sc[0][r];
    #pragma unroll
    for (int gt = 1; gt < 5; ++gt) mx = fmaxf(mx, sc[gt][r]);
    mx = fmaxf(mx, __shfl_xor(mx, 1));
    mx = fmaxf(mx, __shfl_xor(mx, 2));
    mx = fmaxf(mx, __shfl_xor(mx, 4));
    mx = fmaxf(mx, __shfl_xor(mx, 8));
    float s = 0.f;
    #pragma unroll
    for (int gt = 0; gt < 5; ++gt) {
      const float p = __expf(sc[gt][r] - mx);
      sc[gt][r] = p;
      s += p;
    }
    s += __shfl_xor(s, 1);
    s += __shfl_xor(s, 2);
    s += __shfl_xor(s, 4);
    s += __shfl_xor(s, 8);
    const float inv = 1.f / s;
    #pragma unroll
    for (int gt = 0; gt < 5; ++gt) sc[gt][r] *= inv;
  }

  // stash P (f x 96, zero-padded g) into LDS, then read back as A-operand
  for (int i = lane; i < 16 * 16; i += 32)
    pl[i >> 4][80 + (i & 15)] = (_Float16)0.f;
  #pragma unroll
  for (int gt = 0; gt < 5; ++gt) {
    #pragma unroll
    for (int r = 0; r < 8; ++r)
      pl[hg * 8 + r][gt * 16 + nq] = (_Float16)sc[gt][r];
  }

  v16h ap[3];
  #pragma unroll
  for (int kb = 0; kb < 3; ++kb) ap[kb] = load_a16(&pl[nq][0] + kb * 32, hg);

  const int b = bh >> 3, hh = bh & 7;
  for (int tt = 0; tt < T_ / 16; ++tt) {
    v8f o = vzero8();
    const _Float16* vrow = V16 + ((size_t)bh * T_ + tt * 16 + nq) * FP_;
    #pragma unroll
    for (int kb = 0; kb < 3; ++kb)
      o = wmma16(ap[kb], load_b16(vrow + kb * 32, hg), o);
    const int t = tt * 16 + nq;
    float* op = attn + (((size_t)(b * 2 * H_ + H_ + hh) * T_ + t) * F_) + ftile * 16 + hg * 8;
    #pragma unroll
    for (int r = 0; r < 8; ++r) op[r] = o[r];
  }
}

// ---------------------------------------------------- final BN + ReLU -------
__global__ __launch_bounds__(256)
void k_bn_apply(float* __restrict__ Y, const float* __restrict__ stats) {
  const size_t gid = (size_t)blockIdx.x * 256 + threadIdx.x;  // B_*E_*T_*F_
  const int c = (int)((gid / TF_) % E_);
  Y[gid] = fmaxf(Y[gid] * stats[2 * c] + stats[2 * c + 1], 0.f);
}

// -------------------------------------------------------------- launcher ----
static inline void* ws_take(char*& p, size_t bytes) {
  void* r = (void*)p;
  p += (bytes + 255) & ~(size_t)255;
  return r;
}

extern "C" void kernel_launch(void* const* d_in, const int* in_sizes, int n_in,
                              void* d_out, int out_size, void* d_ws, size_t ws_size,
                              hipStream_t stream) {
  (void)in_sizes; (void)n_in; (void)out_size; (void)ws_size;
  const float* x    = (const float*)d_in[0];
  const float* w_in = (const float*)d_in[1];
  const float* b_in = (const float*)d_in[2];
  const float* w_out = (const float*)d_in[3];
  const float* b_out = (const float*)d_in[4];
  const float* gq = (const float*)d_in[5];
  const float* bq = (const float*)d_in[6];
  const float* gk = (const float*)d_in[7];
  const float* bk = (const float*)d_in[8];
  const float* gv = (const float*)d_in[9];
  const float* bv = (const float*)d_in[10];
  const float* go = (const float*)d_in[11];
  const float* bo = (const float*)d_in[12];

  float* out = (float*)d_out;                                // (B,E,T,F)
  float* aw  = out + (size_t)B_ * E_ * T_ * F_;              // (B,T,T)

  char* p = (char*)d_ws;
  float*    qkv   = (float*)   ws_take(p, (size_t)B_ * C3_ * TF_ * 4);
  _Float16* Q16   = (_Float16*)ws_take(p, (size_t)BH_ * T_ * FP_ * 2);
  _Float16* K16   = (_Float16*)ws_take(p, (size_t)BH_ * T_ * FP_ * 2);
  _Float16* V16   = (_Float16*)ws_take(p, (size_t)BH_ * T_ * FP_ * 2);
  _Float16* Qt    = (_Float16*)ws_take(p, (size_t)BH_ * FP_ * T_ * 2);
  _Float16* Kt    = (_Float16*)ws_take(p, (size_t)BH_ * FP_ * T_ * 2);
  _Float16* Vt    = (_Float16*)ws_take(p, (size_t)BH_ * FP_ * T_ * 2);
  float*    attn  = (float*)   ws_take(p, (size_t)B_ * 2 * H_ * TF_ * 4);
  float*    ml    = (float*)   ws_take(p, (size_t)BH_ * T_ * 2 * 4);
  float*    pS_in = (float*)   ws_take(p, (size_t)C3_ * B_ * NBLK_ * 4);
  float*    pQ_in = (float*)   ws_take(p, (size_t)C3_ * B_ * NBLK_ * 4);
  float*    pS_out= (float*)   ws_take(p, (size_t)E_ * B_ * NBLK_ * 4);
  float*    pQ_out= (float*)   ws_take(p, (size_t)E_ * B_ * NBLK_ * 4);
  float*    st_in = (float*)   ws_take(p, 2 * C3_ * 4);
  float*    st_out= (float*)   ws_take(p, 2 * E_ * 4);

  k_conv3x3_stats<<<dim3(NBLK_, B_ * C3_), 256, 0, stream>>>(
      x, w_in, b_in, qkv, pS_in, pQ_in, E_, C3_, /*scaleLim=*/H_);
  k_bn_finalize_in<<<C3_, 256, 0, stream>>>(pS_in, pQ_in, gq, bq, gk, bk, gv, bv, st_in);
  k_pack<<<(BH_ * T_ * FP_) / 256, 256, 0, stream>>>(qkv, st_in, Q16, K16, V16, Qt, Kt, Vt);

  k_time_attn<<<dim3(BH_, 16), 128, 0, stream>>>(Q16, K16, Vt, attn, ml);
  k_aw<<<dim3(B_, 16), 128, 0, stream>>>(Q16, K16, ml, aw);
  k_freq_attn<<<dim3(BH_, 5), 32, 0, stream>>>(Qt, Kt, V16, attn);

  k_conv3x3_stats<<<dim3(NBLK_, B_ * E_), 256, 0, stream>>>(
      attn, w_out, b_out, out, pS_out, pQ_out, 2 * H_, E_, /*scaleLim=*/0);
  k_bn_finalize_out<<<E_, 256, 0, stream>>>(pS_out, pQ_out, go, bo, st_out);
  k_bn_apply<<<(B_ * E_ * TF_) / 256, 256, 0, stream>>>(out, st_out);
}
